// EDTNN_ViolenceDetection_15384572854802
// MI455X (gfx1250) — compile-verified, matrix operand order checked by reading, not run on hardware
//
#include <hip/hip_runtime.h>
#include <hip/hip_bf16.h>
#include <math.h>

// Problem constants (from reference)
#define B_SZ   2048
#define IN_F   4096
#define OUT_F  4096
#define NNODE  64
#define DEG    4
#define OPN    (OUT_F / NNODE)   // 64
#define NEDGE  (NNODE * DEG)     // 256

typedef __attribute__((ext_vector_type(16))) __bf16 v16bf;
typedef __attribute__((ext_vector_type(8)))  __bf16 v8bf;
typedef __attribute__((ext_vector_type(8)))  float  v8f;
typedef int v4i __attribute__((vector_size(16)));   // matches builtin param type

#define ASG __attribute__((address_space(1)))
#define ASL __attribute__((address_space(3)))

#if __has_builtin(__builtin_amdgcn_global_load_async_to_lds_b128)
#define HAS_ASYNC 1
#endif

#ifdef HAS_ASYNC
# if __has_builtin(__builtin_amdgcn_s_wait_asynccnt)
#  define WAIT_ASYNC(n) __builtin_amdgcn_s_wait_asynccnt(n)
# else
#  define WAIT_ASYNC(n) asm volatile("s_wait_asynccnt " #n ::: "memory")
# endif
#else
# define WAIT_ASYNC(n) ((void)0)
#endif

// ---------------------------------------------------------------------------
// Kernel 1: per-edge scale -> per-dst-node (src, scale) lists.
// scale_e = (1 + eps*cos(phi)) / (1 + tau_s*tau_d)
// ---------------------------------------------------------------------------
__global__ void edge_prep(const float* __restrict__ ecoeff,
                          const float* __restrict__ phase,
                          const float* __restrict__ tension,
                          const int*   __restrict__ src,
                          const int*   __restrict__ dst,
                          int*   __restrict__ srcList,    // [NNODE][DEG]
                          float* __restrict__ scaleList)  // [NNODE][DEG]
{
    __shared__ int cnt[NNODE];
    int t = threadIdx.x;
    if (t < NNODE) cnt[t] = 0;
    __syncthreads();
    if (t < NEDGE) {
        int s = src[t], d = dst[t];
        float eps = ecoeff[s * NNODE + d];
        float phi = phase[s * NNODE + d];
        float tau = tension[s] * tension[d];
        float sc  = (1.0f + eps * cosf(phi)) / (1.0f + tau);
        int slot = atomicAdd(&cnt[d], 1);
        if (slot < DEG) {
            srcList[d * DEG + slot]   = s;
            scaleList[d * DEG + slot] = sc;
        }
    }
}

// ---------------------------------------------------------------------------
// Kernel 2: fold entanglement mixing into weights, emit bf16.
// Wp[n*OPN+o, k] = W[n*OPN+o, k] + 0.5 * sum_t scale[n][t] * W[src[n][t]*OPN+o, k]
// ---------------------------------------------------------------------------
__global__ void fold_weights(const float* __restrict__ W,
                             const int*   __restrict__ srcList,
                             const float* __restrict__ scaleList,
                             __bf16*      __restrict__ Wp)
{
    long gid  = (long)blockIdx.x * blockDim.x + threadIdx.x;
    long base = gid * 8;
    int  j = (int)(base / IN_F);
    int  k = (int)(base % IN_F);
    int  n = j >> 6;
    int  o = j & (OPN - 1);

    int   s0 = srcList[n * DEG + 0], s1 = srcList[n * DEG + 1];
    int   s2 = srcList[n * DEG + 2], s3 = srcList[n * DEG + 3];
    float c0 = 0.5f * scaleList[n * DEG + 0];
    float c1 = 0.5f * scaleList[n * DEG + 1];
    float c2 = 0.5f * scaleList[n * DEG + 2];
    float c3 = 0.5f * scaleList[n * DEG + 3];

    const float* wr = W + (long)j * IN_F + k;
    const float* r0 = W + ((long)(s0 * OPN + o)) * IN_F + k;
    const float* r1 = W + ((long)(s1 * OPN + o)) * IN_F + k;
    const float* r2 = W + ((long)(s2 * OPN + o)) * IN_F + k;
    const float* r3 = W + ((long)(s3 * OPN + o)) * IN_F + k;

    v8bf outv;
#pragma unroll
    for (int i = 0; i < 8; ++i) {
        float v = wr[i] + c0 * r0[i] + c1 * r1[i] + c2 * r2[i] + c3 * r3[i];
        outv[i] = (__bf16)v;
    }
    *reinterpret_cast<v8bf*>(Wp + base) = outv;
}

// ---------------------------------------------------------------------------
// Kernel 3: convert x (fp32) to bf16.
// ---------------------------------------------------------------------------
__global__ void convert_x(const float* __restrict__ x, __bf16* __restrict__ xb)
{
    long base = ((long)blockIdx.x * blockDim.x + threadIdx.x) * 8;
    v8bf outv;
#pragma unroll
    for (int i = 0; i < 8; ++i) outv[i] = (__bf16)x[base + i];
    *reinterpret_cast<v8bf*>(xb + base) = outv;
}

// ---------------------------------------------------------------------------
// Kernel 4: C = Xb * Wp^T with async global->LDS double-buffered staging.
//
// Block: 256 threads = 8 wave32s arranged 2(M) x 4(N).
// Block tile 128(M) x 256(N); wave tile 64x64 = 4x4 WMMA accs.
// K-step 32 (one v_wmma_f32_16x16x32_bf16 K-chunk per stage).
//
// LDS per stage: A 128 rows x 64B (padded to 80B) = 10240B
//                B 256 rows x 64B (padded to 80B) = 20480B
// Double-buffered total: 61440B. The 80B row stride makes the 16 row reads of
// a fragment land on disjoint 4-bank windows (20*r mod 64 covers 0..60 step 4).
//
// Per stage each thread issues 6 async b128 copies (ASYNCcnt += 6).
// Pipeline: issue(s+1); s_wait_asynccnt<=6 (in-order => stage s landed);
// barrier; consume (16 ds_load_b128 + 16 wmma per wave); barrier.
// ---------------------------------------------------------------------------
#define A_STAGE_BYTES (128 * 80)   // 10240
#define B_STAGE_BYTES (256 * 80)   // 20480
#define STAGE_BYTES   (A_STAGE_BYTES + B_STAGE_BYTES)

__device__ __forceinline__ void async_copy16(const void* g, void* l)
{
#ifdef HAS_ASYNC
    __builtin_amdgcn_global_load_async_to_lds_b128((ASG v4i*)g, (ASL v4i*)l, 0, 0);
#else
    *reinterpret_cast<int4*>(l) = *reinterpret_cast<const int4*>(g);
#endif
}

// gA/gB point at (row 0, current k) in bytes; row stride = IN_F*2 bytes.
__device__ __forceinline__ void stage_copy(const char* gA, const char* gB,
                                           char* lA, char* lB, int tid)
{
#pragma unroll
    for (int t = 0; t < 2; ++t) {           // A: 512 chunks of 16B
        int c = tid + t * 256;
        int row = c >> 2, cx = c & 3;
        async_copy16(gA + (long)row * (IN_F * 2) + cx * 16,
                     lA + row * 80 + cx * 16);
    }
#pragma unroll
    for (int t = 0; t < 4; ++t) {           // B: 1024 chunks of 16B
        int c = tid + t * 256;
        int row = c >> 2, cx = c & 3;
        async_copy16(gB + (long)row * (IN_F * 2) + cx * 16,
                     lB + row * 80 + cx * 16);
    }
}

// Fragment from an LDS tile row: p = rowbase + (lane>=16 ? 16 : 0).
// lanes 0-15: K0..7 @ +0,  K16..23 @ +32 ; lanes 16-31: K8..15, K24..31.
__device__ __forceinline__ v16bf lds_frag(const char* p)
{
    v8bf lo = *reinterpret_cast<const v8bf*>(p);
    v8bf hi = *reinterpret_cast<const v8bf*>(p + 32);
    v16bf r;
#pragma unroll
    for (int i = 0; i < 8; ++i) { r[i] = lo[i]; r[i + 8] = hi[i]; }
    return r;
}

__global__ __launch_bounds__(256)
void gemm_bf16_wmma(const __bf16* __restrict__ A,   // [B_SZ, IN_F]
                    const __bf16* __restrict__ Bw,  // [OUT_F, IN_F]
                    float*        __restrict__ C)   // [B_SZ, OUT_F]
{
    __shared__ __align__(16) char smem[2][STAGE_BYTES];

    const int tid  = threadIdx.x;
    const int lane = tid & 31;
    const int wave = tid >> 5;
    const int wm   = wave >> 2;          // 0..1
    const int wn   = wave & 3;           // 0..3
    const int mblk = blockIdx.y * 128;
    const int nblk = blockIdx.x * 256;

    const int r   = lane & 15;
    const int sel = (lane >> 4) << 4;    // byte offset 0 or 16 within row

    const char* gA0 = (const char*)(A  + (long)mblk * IN_F);
    const char* gB0 = (const char*)(Bw + (long)nblk * IN_F);

    // LDS row-base byte offsets for this wave's fragments
    int aOff[4], bOff[4];
#pragma unroll
    for (int fm = 0; fm < 4; ++fm) aOff[fm] = (wm * 64 + fm * 16 + r) * 80 + sel;
#pragma unroll
    for (int fn = 0; fn < 4; ++fn) bOff[fn] = (wn * 64 + fn * 16 + r) * 80 + sel;

    v8f acc[4][4];
#pragma unroll
    for (int fm = 0; fm < 4; ++fm)
#pragma unroll
        for (int fn = 0; fn < 4; ++fn)
            acc[fm][fn] = (v8f)(0.0f);

    const int NST = IN_F / 32;           // 128 stages

    // Prologue: stage 0 into buffer 0
    stage_copy(gA0, gB0, smem[0], smem[0] + A_STAGE_BYTES, tid);

    for (int s = 0; s < NST; ++s) {
        const int cur = s & 1;
        if (s + 1 < NST) {
            stage_copy(gA0 + (long)(s + 1) * 64,      // 32 bf16 = 64 bytes of K
                       gB0 + (long)(s + 1) * 64,
                       smem[cur ^ 1], smem[cur ^ 1] + A_STAGE_BYTES, tid);
            WAIT_ASYNC(6);               // stage s done (in-order completion)
        } else {
            WAIT_ASYNC(0);
        }
        __syncthreads();

        const char* As = smem[cur];
        const char* Bs = As + A_STAGE_BYTES;

        v16bf af[4], bfr[4];
#pragma unroll
        for (int fm = 0; fm < 4; ++fm) af[fm]  = lds_frag(As + aOff[fm]);
#pragma unroll
        for (int fn = 0; fn < 4; ++fn) bfr[fn] = lds_frag(Bs + bOff[fn]);

#pragma unroll
        for (int fm = 0; fm < 4; ++fm)
#pragma unroll
            for (int fn = 0; fn < 4; ++fn)
                acc[fm][fn] = __builtin_amdgcn_wmma_f32_16x16x32_bf16(
                    false, af[fm], false, bfr[fn],
                    (short)0, acc[fm][fn], false, false);

        __syncthreads();                 // buffer cur free for stage s+2
    }

    // C/D layout: VGPR v: lanes 0-15 -> M=v, lanes 16-31 -> M=8+v; N = lane&15
    const int rbase = (lane >> 4) << 3;
    const int col   = lane & 15;
#pragma unroll
    for (int fm = 0; fm < 4; ++fm) {
#pragma unroll
        for (int fn = 0; fn < 4; ++fn) {
            float* cp = C + (long)(mblk + wm * 64 + fm * 16 + rbase) * OUT_F
                          + (nblk + wn * 64 + fn * 16 + col);
#pragma unroll
            for (int v = 0; v < 8; ++v)
                cp[(long)v * OUT_F] = acc[fm][fn][v];
        }
    }
}

// ---------------------------------------------------------------------------
// Launch
// ---------------------------------------------------------------------------
extern "C" void kernel_launch(void* const* d_in, const int* in_sizes, int n_in,
                              void* d_out, int out_size, void* d_ws, size_t ws_size,
                              hipStream_t stream)
{
    const float* x       = (const float*)d_in[0];   // [B_SZ, IN_F]
    const float* W       = (const float*)d_in[1];   // [OUT_F, IN_F]
    const float* ecoeff  = (const float*)d_in[2];   // [N, N]
    const float* phase   = (const float*)d_in[3];   // [N, N]
    const float* tension = (const float*)d_in[4];   // [N]
    const int*   src     = (const int*)d_in[5];     // [E]
    const int*   dst     = (const int*)d_in[6];     // [E]
    float*       out     = (float*)d_out;           // [B_SZ, OUT_F]

    char* ws = (char*)d_ws;
    __bf16* xb = (__bf16*)ws;                                          // 16 MB
    __bf16* wp = (__bf16*)(ws + (size_t)B_SZ * IN_F * sizeof(__bf16)); // 32 MB
    char*   tail = ws + (size_t)B_SZ * IN_F * sizeof(__bf16)
                      + (size_t)OUT_F * IN_F * sizeof(__bf16);
    int*    srcList   = (int*)tail;
    float*  scaleList = (float*)(srcList + NNODE * DEG);

    edge_prep<<<1, 256, 0, stream>>>(ecoeff, phase, tension, src, dst,
                                     srcList, scaleList);
    {
        long total = (long)OUT_F * IN_F / 8;
        fold_weights<<<(int)(total / 256), 256, 0, stream>>>(W, srcList, scaleList, wp);
    }
    {
        long total = (long)B_SZ * IN_F / 8;
        convert_x<<<(int)(total / 256), 256, 0, stream>>>(x, xb);
    }
    {
        dim3 grid(OUT_F / 256, B_SZ / 128);   // (16, 16)
        gemm_bf16_wmma<<<grid, 256, 0, stream>>>(xb, wp, out);
    }
}
